// Attention_69243462746479
// MI455X (gfx1250) — matrix operand order repaired and checked by reading, not compiled, not run on hardware
//
#include <hip/hip_runtime.h>
#include <hip/hip_bf16.h>

// ---------------------------------------------------------------------------
// Attention (B=2, S=2048, D=2048, NH=16, HD=128), fp32 in/out.
// bf16 WMMA (v_wmma_f32_16x16x32_bf16) for all matmuls; flash-style attention.
// Async staging of the shared K tile into LDS via
// __builtin_amdgcn_global_load_async_to_lds_b128 (signature confirmed by the
// round-2 diagnostic: (v4i __device__*, v4i __shared__*, imm, imm)), plus
// global_prefetch hints for next tiles.
// Workspace requirement: 4 * 4096*2048 bf16 buffers = ~67 MB.
// ---------------------------------------------------------------------------

typedef __attribute__((ext_vector_type(16))) __bf16 v16bf;
typedef __attribute__((ext_vector_type(8)))  float  v8f;
typedef int v4i_vec __attribute__((vector_size(16)));

#define ATT_B  2
#define ATT_S  2048
#define ATT_D  2048
#define ATT_NH 16
#define ATT_HD 128

// ---- async global->LDS copy (16 bytes), guarded ----------------------------
__device__ inline void async_copy16(const __bf16* g, __bf16* l) {
#if defined(__AMDGCN__) && __has_builtin(__builtin_amdgcn_global_load_async_to_lds_b128)
  __builtin_amdgcn_global_load_async_to_lds_b128(
      (__attribute__((address_space(1))) v4i_vec*)g,
      (__attribute__((address_space(3))) v4i_vec*)l, 0, 0);
#else
  *reinterpret_cast<uint4*>(l) = *reinterpret_cast<const uint4*>(g);
#endif
}
__device__ inline void async_join() {
#if defined(__AMDGCN__) && __has_builtin(__builtin_amdgcn_global_load_async_to_lds_b128)
#if __has_builtin(__builtin_amdgcn_s_wait_asynccnt)
  __builtin_amdgcn_s_wait_asynccnt(0);
#else
  asm volatile("s_wait_asynccnt 0x0" ::: "memory");
#endif
#endif
}

// ---- WMMA fragment loaders (wave32, 16x16x32 bf16) -------------------------
// A fragment 16x32 (M x K): lane m = lane&15, half = lane>>4.
//   elems[0..7]  = A[m][half*8 + i]
//   elems[8..15] = A[m][16 + half*8 + i]
__device__ inline v16bf load_a_frag(const __bf16* base, int ld) {
  const int lane = threadIdx.x & 31;
  const int m = lane & 15, half = lane >> 4;
  const __bf16* p0 = base + (size_t)m * ld + half * 8;
  union { v16bf v; uint4 q[2]; } u;
  u.q[0] = *reinterpret_cast<const uint4*>(p0);
  u.q[1] = *reinterpret_cast<const uint4*>(p0 + 16);
  return u.v;
}

// B fragment 32x16 (K x N) where B[k][n] is stored row-major by n with
// contiguous k (i.e. source row n holds k=0..31 contiguously):
//   lane n = lane&15, kb = (lane>>4)*8; elems[i] = src[n*ld + kb + i] (i<8),
//   elems[8+i] = src[n*ld + 16 + kb + i]
__device__ inline v16bf load_b_frag(const __bf16* base, int ld) {
  const int lane = threadIdx.x & 31;
  const int n = lane & 15, kb = (lane >> 4) * 8;
  const __bf16* p = base + (size_t)n * ld + kb;
  union { v16bf v; uint4 q[2]; } u;
  u.q[0] = *reinterpret_cast<const uint4*>(p);
  u.q[1] = *reinterpret_cast<const uint4*>(p + 16);
  return u.v;
}

// ---- LDS staging with on-the-fly f32 -> bf16 conversion --------------------
__device__ inline void stage16(const float* g, __bf16* l) {
  #pragma unroll
  for (int i = 0; i < 4; ++i) {
    float4 v = reinterpret_cast<const float4*>(g)[i];
    l[i * 4 + 0] = (__bf16)v.x;
    l[i * 4 + 1] = (__bf16)v.y;
    l[i * 4 + 2] = (__bf16)v.z;
    l[i * 4 + 3] = (__bf16)v.w;
  }
}
__device__ inline void stage16(const __bf16* g, __bf16* l) {
  reinterpret_cast<uint4*>(l)[0] = reinterpret_cast<const uint4*>(g)[0];
  reinterpret_cast<uint4*>(l)[1] = reinterpret_cast<const uint4*>(g)[1];
}

// ---- GEMM: Y[M,N] = X[M,K] * W[N,K]^T  (both inputs k-contiguous) ----------
// 128x128 block tile, BK=32, 8 waves in 2x4, each wave 64x32 (4x2 WMMA accs).
template <typename XT, typename WT, typename OT>
__global__ __launch_bounds__(256)
void gemm_xwt(const XT* __restrict__ X, const WT* __restrict__ W,
              OT* __restrict__ Y, int M, int N, int K) {
  __shared__ __align__(16) __bf16 sX[128 * 32];
  __shared__ __align__(16) __bf16 sW[128 * 32];

  const int m0 = blockIdx.y * 128;
  const int n0 = blockIdx.x * 128;
  const int t    = threadIdx.x;
  const int lane = t & 31;
  const int wid  = t >> 5;
  const int wm = (wid >> 2) * 64;   // wave row offset (2 rows of waves)
  const int wn = (wid & 3) * 32;    // wave col offset (4 cols of waves)

  v8f acc[4][2] = {};

  const int r  = t >> 1;            // staging: row within 128-row tile
  const int c0 = (t & 1) * 16;      // staging: 16-element half-row

  for (int kk = 0; kk < K; kk += 32) {
    stage16(X + (size_t)(m0 + r) * K + kk + c0, sX + r * 32 + c0);
    stage16(W + (size_t)(n0 + r) * K + kk + c0, sW + r * 32 + c0);
    if (kk + 32 < K) {  // prefetch next k-slice (global_prefetch_b8)
      __builtin_prefetch(X + (size_t)(m0 + r) * K + kk + 32 + c0, 0, 1);
      __builtin_prefetch(W + (size_t)(n0 + r) * K + kk + 32 + c0, 0, 1);
    }
    __syncthreads();

    v16bf a[4], b[2];
    #pragma unroll
    for (int i = 0; i < 4; ++i) a[i] = load_a_frag(sX + (wm + i * 16) * 32, 32);
    #pragma unroll
    for (int j = 0; j < 2; ++j) b[j] = load_b_frag(sW + (wn + j * 16) * 32, 32);

    #pragma unroll
    for (int i = 0; i < 4; ++i)
      #pragma unroll
      for (int j = 0; j < 2; ++j)
        acc[i][j] = __builtin_amdgcn_wmma_f32_16x16x32_bf16(
            false, a[i], false, b[j], (short)0, acc[i][j], false, false);
    __syncthreads();
  }

  const int half = lane >> 4, nl = lane & 15;
  #pragma unroll
  for (int i = 0; i < 4; ++i)
    #pragma unroll
    for (int j = 0; j < 2; ++j)
      #pragma unroll
      for (int e = 0; e < 8; ++e) {
        const int row = m0 + wm + i * 16 + e + 8 * half;
        const int col = n0 + wn + j * 16 + nl;
        Y[(size_t)row * N + col] = (OT)acc[i][j][e];
      }
}

// ---- RoPE on Q and K (bf16 in-place) ---------------------------------------
__global__ void rope_kernel(__bf16* __restrict__ q, __bf16* __restrict__ k,
                            const float* __restrict__ fcos,
                            const float* __restrict__ fsin) {
  const size_t idx = (size_t)blockIdx.x * blockDim.x + threadIdx.x;
  if (idx >= (size_t)ATT_B * ATT_S * ATT_NH * (ATT_HD / 2)) return;
  const int i   = idx & 63;          // rotary pair index within head
  const int h   = (idx >> 6) & 15;
  const int row = (int)(idx >> 10);  // b*S + s
  const int s   = row & (ATT_S - 1);
  const float c  = fcos[s * 64 + i];
  const float sn = fsin[s * 64 + i];
  const size_t p = (size_t)row * ATT_D + h * ATT_HD + 2 * i;
  float q0 = (float)q[p], q1 = (float)q[p + 1];
  q[p]     = (__bf16)(q0 * c - q1 * sn);
  q[p + 1] = (__bf16)(q0 * sn + q1 * c);
  float k0 = (float)k[p], k1 = (float)k[p + 1];
  k[p]     = (__bf16)(k0 * c - k1 * sn);
  k[p + 1] = (__bf16)(k0 * sn + k1 * c);
}

// ---- V transpose: (b,s,h,d) -> (b,h,d,s) so P@V B-frags are k-contiguous ---
__global__ void transpose_v(const __bf16* __restrict__ v,
                            __bf16* __restrict__ vt) {
  const size_t idx = (size_t)blockIdx.x * blockDim.x + threadIdx.x;
  if (idx >= (size_t)ATT_B * ATT_S * ATT_D) return;
  const int d   = idx & 127;
  const int h   = (idx >> 7) & 15;
  const size_t row = idx >> 11;       // b*S + s
  const int s = (int)(row & (ATT_S - 1));
  const int b = (int)(row >> 11);
  vt[(((size_t)b * ATT_NH + h) * ATT_HD + d) * ATT_S + s] = v[idx];
}

// ---- Flash attention: one block per (b, h, 128 q-rows); wave = 16 q-rows ---
// K tile (64 keys x 128 d, shared by all 8 waves) is staged into LDS via the
// async global->LDS path once per key block.
__global__ __launch_bounds__(256)
void flash_attn(const __bf16* __restrict__ q, const __bf16* __restrict__ k,
                const __bf16* __restrict__ vt, __bf16* __restrict__ ctx) {
  constexpr int S = ATT_S, D = ATT_D, HD = ATT_HD, NH = ATT_NH;
  __shared__ __align__(16) __bf16 sP[8][16 * 64];  // per-wave P staging (16KB)
  __shared__ __align__(16) __bf16 sK[64 * 128];    // shared K tile     (16KB)

  const int qblk = blockIdx.x, h = blockIdx.y, b = blockIdx.z;
  const int t = threadIdx.x, lane = t & 31, wid = t >> 5;
  const int half = lane >> 4, nl = lane & 15;
  const int q0 = qblk * 128 + wid * 16;

  const __bf16* qBase  = q  + (size_t)(b * S + q0) * D + h * HD;
  const __bf16* kBase  = k  + (size_t)(b * S) * D + h * HD;
  const __bf16* vtBase = vt + (size_t)(b * NH + h) * HD * S;

  // Q fragments for this wave's 16 rows (4 chunks along HD=128), kept live.
  v16bf aq[4];
  #pragma unroll
  for (int j = 0; j < 4; ++j) aq[j] = load_a_frag(qBase + j * 32, D);

  v8f oacc[8] = {};                 // 16 x 128 output accumulator
  float mrow[8], lrow[8];
  #pragma unroll
  for (int i = 0; i < 8; ++i) { mrow[i] = -3.0e38f; lrow[i] = 0.0f; }

  const float scale = 0.0883883476483184f;  // 1/sqrt(128)
  const int nkb = (qblk + 1) * 2;  // causal: key blocks of 64 up to diagonal

  for (int kb = 0; kb < nkb; ++kb) {
    const int key0 = kb * 64;

    // ---- stage K tile 64x128 bf16 into LDS (async; 4 x b128 per thread) ----
    {
      #pragma unroll
      for (int j = 0; j < 4; ++j) {
        const int chunk = t * 4 + j;        // 16B chunks; 16 chunks per row
        const int krow  = chunk >> 4;
        const int celt  = (chunk & 15) * 8; // element offset within row
        async_copy16(kBase + (size_t)(key0 + krow) * D + celt,
                     sK + krow * 128 + celt);
      }
      async_join();
    }
    if (kb + 1 < nkb) {  // prefetch next key block of K and V^T
      __builtin_prefetch(kBase + (size_t)(key0 + 64 + (t >> 2)) * D, 0, 1);
      __builtin_prefetch(vtBase + (size_t)(t >> 1) * S + key0 + 64, 0, 1);
    }
    __syncthreads();

    // ---- S = Q K^T for 16x64 tile (4 key sub-tiles, 4 d-chunks each) ----
    v8f sacc[4] = {};
    #pragma unroll
    for (int c = 0; c < 4; ++c)
      #pragma unroll
      for (int j = 0; j < 4; ++j) {
        v16bf bk = load_b_frag(sK + (c * 16) * 128 + j * 32, 128);
        sacc[c] = __builtin_amdgcn_wmma_f32_16x16x32_bf16(
            false, aq[j], false, bk, (short)0, sacc[c], false, false);
      }

    // ---- causal mask + scale + running row max ----
    float rmax[8];
    #pragma unroll
    for (int i = 0; i < 8; ++i) rmax[i] = mrow[i];
    #pragma unroll
    for (int c = 0; c < 4; ++c)
      #pragma unroll
      for (int i = 0; i < 8; ++i) {
        const int qrow = q0 + i + 8 * half;
        const int key  = key0 + c * 16 + nl;
        float sv = sacc[c][i] * scale;
        sv = (key <= qrow) ? sv : -3.0e38f;
        sacc[c][i] = sv;
        rmax[i] = fmaxf(rmax[i], sv);
      }
    #pragma unroll
    for (int i = 0; i < 8; ++i) {   // max across the 16-lane half
      float v = rmax[i];
      v = fmaxf(v, __shfl_xor(v, 1, 32));
      v = fmaxf(v, __shfl_xor(v, 2, 32));
      v = fmaxf(v, __shfl_xor(v, 4, 32));
      v = fmaxf(v, __shfl_xor(v, 8, 32));
      rmax[i] = v;
    }

    // ---- exp, row sums, P -> LDS (bf16) ----
    float rsum[8];
    #pragma unroll
    for (int i = 0; i < 8; ++i) rsum[i] = 0.0f;
    #pragma unroll
    for (int c = 0; c < 4; ++c)
      #pragma unroll
      for (int i = 0; i < 8; ++i) {
        const float p = __expf(sacc[c][i] - rmax[i]);
        rsum[i] += p;
        sP[wid][(i + 8 * half) * 64 + c * 16 + nl] = (__bf16)p;
      }
    #pragma unroll
    for (int i = 0; i < 8; ++i) {
      float v = rsum[i];
      v += __shfl_xor(v, 1, 32);
      v += __shfl_xor(v, 2, 32);
      v += __shfl_xor(v, 4, 32);
      v += __shfl_xor(v, 8, 32);
      const float alpha = __expf(mrow[i] - rmax[i]);
      lrow[i] = lrow[i] * alpha + v;
      mrow[i] = rmax[i];
      #pragma unroll
      for (int dt = 0; dt < 8; ++dt) oacc[dt][i] *= alpha;
    }
    __syncthreads();  // P visible (all waves run identical iteration count)

    // ---- O += P V  (A from LDS, B from pre-transposed V, k-contiguous) ----
    v16bf ap[2];
    ap[0] = load_a_frag(&sP[wid][0], 64);
    ap[1] = load_a_frag(&sP[wid][32], 64);
    #pragma unroll
    for (int dt = 0; dt < 8; ++dt)
      #pragma unroll
      for (int tt = 0; tt < 2; ++tt) {
        v16bf bv = load_b_frag(vtBase + (size_t)(dt * 16) * S + key0 + tt * 32, S);
        oacc[dt] = __builtin_amdgcn_wmma_f32_16x16x32_bf16(
            false, ap[tt], false, bv, (short)0, oacc[dt], false, false);
      }
    __syncthreads();  // all reads of sK/sP done before next-iter staging
  }

  // ---- finalize: divide by l, write bf16 context ----
  #pragma unroll
  for (int dt = 0; dt < 8; ++dt)
    #pragma unroll
    for (int e = 0; e < 8; ++e) {
      const int qrow = q0 + e + 8 * half;
      const int d    = dt * 16 + nl;
      ctx[(size_t)(b * S + qrow) * D + h * HD + d] =
          (__bf16)(oacc[dt][e] / lrow[e]);
    }
}

// ---------------------------------------------------------------------------
extern "C" void kernel_launch(void* const* d_in, const int* in_sizes, int n_in,
                              void* d_out, int out_size, void* d_ws,
                              size_t ws_size, hipStream_t stream) {
  const float* x  = (const float*)d_in[0];
  const float* wq = (const float*)d_in[1];
  const float* wk = (const float*)d_in[2];
  const float* wv = (const float*)d_in[3];
  const float* wo = (const float*)d_in[4];
  const float* fc = (const float*)d_in[5];
  const float* fs = (const float*)d_in[6];
  // d_in[7] (mask) unused: causal mask applied analytically.
  float* out = (float*)d_out;

  constexpr size_t MROWS = (size_t)ATT_B * ATT_S;   // 4096
  constexpr size_t NELM  = MROWS * ATT_D;           // 8,388,608

  __bf16* qb  = (__bf16*)d_ws;
  __bf16* kb  = qb + NELM;
  __bf16* vb  = kb + NELM;
  __bf16* vtb = vb + NELM;
  __bf16* ctx = vb;   // vb is dead after transpose_v; reuse for context

  dim3 gblk(ATT_D / 128, (unsigned)(MROWS / 128));  // 16 x 32

  gemm_xwt<float, float, __bf16><<<gblk, 256, 0, stream>>>(
      x, wq, qb, (int)MROWS, ATT_D, ATT_D);
  gemm_xwt<float, float, __bf16><<<gblk, 256, 0, stream>>>(
      x, wk, kb, (int)MROWS, ATT_D, ATT_D);
  gemm_xwt<float, float, __bf16><<<gblk, 256, 0, stream>>>(
      x, wv, vb, (int)MROWS, ATT_D, ATT_D);

  rope_kernel<<<(ATT_B * ATT_S * ATT_NH * (ATT_HD / 2)) / 256, 256, 0,
                stream>>>(qb, kb, fc, fs);
  transpose_v<<<(unsigned)(NELM / 256), 256, 0, stream>>>(vb, vtb);

  dim3 gfa(ATT_S / 128, ATT_NH, ATT_B);
  flash_attn<<<gfa, 256, 0, stream>>>(qb, kb, vtb, ctx);

  gemm_xwt<__bf16, float, float><<<gblk, 256, 0, stream>>>(
      ctx, wo, out, (int)MROWS, ATT_D, ATT_D);
}
